// LzqGatedSparseConvUNet_89232240542138
// MI455X (gfx1250) — matrix operand order repaired and checked by reading, not compiled, not run on hardware
//
#include <hip/hip_runtime.h>
#include <math.h>

// ---------------- types ----------------
typedef __attribute__((ext_vector_type(16))) _Float16 v16h;
typedef __attribute__((ext_vector_type(8)))  float    v8f;

struct __align__(16) U128 { unsigned x, y, z, w; };
union Frag { v16h v; U128 u[2]; };

#define LEAKK 0.1f
#define EPSS 1e-5f

// ---------------- small utility kernels ----------------
__global__ void __launch_bounds__(256) tof16_kernel(const float* __restrict__ X,
                                                    _Float16* __restrict__ Y, size_t n) {
  size_t i = (size_t)blockIdx.x * 256 + threadIdx.x;
  if (i < n) Y[i] = (_Float16)X[i];
}

__global__ void __launch_bounds__(256) zero_kernel(float* __restrict__ p, size_t n) {
  size_t i = (size_t)blockIdx.x * 256 + threadIdx.x;
  if (i < n) p[i] = 0.0f;
}

// W (K, cin, cout2) f32 -> Wt (K, cout2, cin) f16   (B-matrix friendly layout)
__global__ void __launch_bounds__(256) wtrans_kernel(const float* __restrict__ W,
                                                     _Float16* __restrict__ Wt,
                                                     int K, int cin, int cout2) {
  size_t n = (size_t)K * cin * cout2;
  size_t i = (size_t)blockIdx.x * 256 + threadIdx.x;
  if (i >= n) return;
  int k  = (int)(i / ((size_t)cin * cout2));
  int r  = (int)(i - (size_t)k * cin * cout2);
  int ci = r / cout2;
  int co = r - ci * cout2;
  Wt[((size_t)k * cout2 + co) * cin + ci] = (_Float16)W[i];
}

// ---------------- core: gather + WMMA submanifold conv ----------------
// Y[N, cout2] = sum_k gather(X, nbr[:,k]) @ W[k]   (bias cancelled by BN)
// block: MT*16 output rows, 256 threads (8 waves).
// TPW = cout2/128 cout-tiles per wave; each B fragment reused MT times.
template <int TPW, int MT>
__global__ void __launch_bounds__(256)
subm_wmma_kernel(const _Float16* __restrict__ X,   // N x cin (f16)
                 const _Float16* __restrict__ Wt,  // 27 x cout2 x cin (f16)
                 const int* __restrict__ nbr,      // N x 27
                 float* __restrict__ Y,            // N x cout2 (f32 pre-BN)
                 int N, int cin, int cout2) {
  extern __shared__ char smem[];
  _Float16* sA = (_Float16*)smem;                       // (MT*16) x cin
  int* sIdx = (int*)(smem + (size_t)MT * 16 * cin * sizeof(_Float16)); // MT*16 ints

  const int m0    = blockIdx.x * (MT * 16);
  const int wave  = threadIdx.x >> 5;
  const int lane  = threadIdx.x & 31;
  const int row16 = lane & 15;
  const int hi    = lane >> 4;

  v8f acc[MT][TPW];
#pragma unroll
  for (int mt = 0; mt < MT; ++mt)
#pragma unroll
    for (int t = 0; t < TPW; ++t) {
      v8f z = {0.f, 0.f, 0.f, 0.f, 0.f, 0.f, 0.f, 0.f};
      acc[mt][t] = z;
    }

  const int chunksPerRow = cin >> 3;              // 8 f16 per 16B chunk
  const int totalChunks  = chunksPerRow * 16 * MT;

  for (int k = 0; k < 27; ++k) {
    if (threadIdx.x < MT * 16) {
      int r = m0 + threadIdx.x;
      sIdx[threadIdx.x] = (r < N) ? nbr[(size_t)r * 27 + k] : -1;
    }
    __syncthreads();
    // cooperative gather of MT*16 rows into LDS via CDNA5 async-to-LDS path;
    // rows with no neighbor get plain ds_store zero-fill (disjoint ranges).
    for (int c = threadIdx.x; c < totalChunks; c += 256) {
      int r = c / chunksPerRow;
      int e = (c - r * chunksPerRow) << 3;
      _Float16* dst = sA + (size_t)r * cin + e;
      int idx = sIdx[r];
      if (idx >= 0) {
        const _Float16* src = X + (size_t)idx * cin + e;
        unsigned ldsOff = (unsigned)(size_t)dst;   // LDS byte offset (low 32 bits)
        asm volatile("global_load_async_to_lds_b128 %0, %1, off"
                     :: "v"(ldsOff), "v"(src) : "memory");
      } else {
        U128 z = {0u, 0u, 0u, 0u};
        *(U128*)dst = z;
      }
    }
    asm volatile("s_wait_asynccnt 0" ::: "memory");
    __syncthreads();

    const _Float16* wk = Wt + (size_t)k * cout2 * cin;
    if (k + 1 < 27)  // pull next tap's weights toward L2 (global_prefetch_b8)
      __builtin_prefetch(wk + (size_t)cout2 * cin, 0, 3);

    for (int kk = 0; kk < cin; kk += 32) {
      // A fragments per ISA 16-bit A 16x32 layout
      Frag a[MT];
#pragma unroll
      for (int mt = 0; mt < MT; ++mt) {
        const _Float16* ap = sA + (size_t)(mt * 16 + row16) * cin + kk + hi * 8;
        a[mt].u[0] = *(const U128*)(ap);
        a[mt].u[1] = *(const U128*)(ap + 16);
      }
#pragma unroll
      for (int t = 0; t < TPW; ++t) {
        int tile = wave + t * 8;
        int col  = (tile << 4) + row16;
        // B fragment per ISA 16-bit B 32x16 layout (lane-half selects K 0..7,16..23 / 8..15,24..31)
        const _Float16* bp = wk + (size_t)col * cin + kk + hi * 8;
        Frag b;
        b.u[0] = *(const U128*)(bp);
        b.u[1] = *(const U128*)(bp + 16);
#pragma unroll
        for (int mt = 0; mt < MT; ++mt)
          acc[mt][t] = __builtin_amdgcn_wmma_f32_16x16x32_f16(
              false, a[mt].v, false, b.v, (short)0, acc[mt][t], false, false);
      }
    }
    __syncthreads();  // protect sA before next tap overwrites it
  }

  // store: C/D layout — lane&15 = N, VGPR r -> M = r + 8*hi
#pragma unroll
  for (int t = 0; t < TPW; ++t) {
    int n = ((wave + t * 8) << 4) + row16;
#pragma unroll
    for (int mt = 0; mt < MT; ++mt) {
#pragma unroll
      for (int r = 0; r < 8; ++r) {
        int m = m0 + mt * 16 + r + 8 * hi;
        if (m < N) Y[(size_t)m * cout2 + n] = acc[mt][t][r];
      }
    }
  }
}

// ---------------- batch-norm folded stats: scale/shift per column ----------------
__global__ void __launch_bounds__(256)
bn_stats_kernel(const float* __restrict__ Y, int n, int C,
                const float* __restrict__ g, const float* __restrict__ be,
                float* __restrict__ scale, float* __restrict__ shift) {
  __shared__ float ss[256], sq[256];
  int tc = threadIdx.x & 7;
  int tr = threadIdx.x >> 3;
  int c  = blockIdx.x * 8 + tc;
  float s = 0.f, q = 0.f;
  for (int r = tr; r < n; r += 32) {
    float v = Y[(size_t)r * C + c];
    s += v; q += v * v;
  }
  ss[threadIdx.x] = s; sq[threadIdx.x] = q;
  __syncthreads();
  for (int o = 128; o >= 8; o >>= 1) {
    if (threadIdx.x < (unsigned)o) { ss[threadIdx.x] += ss[threadIdx.x + o]; sq[threadIdx.x] += sq[threadIdx.x + o]; }
    __syncthreads();
  }
  if (threadIdx.x < 8) {
    int cc = blockIdx.x * 8 + threadIdx.x;
    float m   = ss[threadIdx.x] / (float)n;
    float var = sq[threadIdx.x] / (float)n - m * m;
    float rr  = rsqrtf(var + EPSS);
    scale[cc] = g[cc] * rr;
    shift[cc] = be[cc] - m * g[cc] * rr;
  }
}

// ---------------- BN apply + gated activation (+ optional residual) ----------------
__global__ void __launch_bounds__(256)
bn_gate_kernel(const float* __restrict__ Y,  // N x 2p (pre-BN)
               const float* __restrict__ scale, const float* __restrict__ shift,
               const float* __restrict__ res,  // N x p or nullptr
               float* __restrict__ xf, _Float16* __restrict__ xh,
               int N, int p) {
  size_t i = (size_t)blockIdx.x * 256 + threadIdx.x;
  if (i >= (size_t)N * p) return;
  int r = (int)(i / p), c = (int)(i - (size_t)r * p);
  float f  = Y[(size_t)r * 2 * p + c]     * scale[c]     + shift[c];
  float gt = Y[(size_t)r * 2 * p + p + c] * scale[p + c] + shift[p + c];
  float lf = f > 0.f ? f : LEAKK * f;
  float sg = 1.f / (1.f + __expf(-gt));
  float o  = lf * sg;
  if (res) o += res[i];
  xf[i] = o;
  xh[i] = (_Float16)o;
}

// ---------------- down conv: per-child GEMV, atomic scatter to parent ----------------
__global__ void __launch_bounds__(256)
down_kernel(const float* __restrict__ X,  // Nc x cin
            const float* __restrict__ W,  // 8 x cin x cout2
            const int* __restrict__ par, const int* __restrict__ off,
            float* __restrict__ Y,        // Np x cout2 (pre-zeroed)
            int Nc, int cin, int cout2) {
  size_t i = (size_t)blockIdx.x * 256 + threadIdx.x;
  if (i >= (size_t)Nc * cout2) return;
  int r = (int)(i / cout2), co = (int)(i - (size_t)r * cout2);
  const float* w = W + ((size_t)off[r] * cin) * cout2 + co;
  const float* x = X + (size_t)r * cin;
  float s = 0.f;
  for (int ci = 0; ci < cin; ++ci) s += x[ci] * w[(size_t)ci * cout2];
  atomicAdd(&Y[(size_t)par[r] * cout2 + co], s);
}

// ---------------- up conv: gather parent row, per-row GEMV ----------------
__global__ void __launch_bounds__(256)
up_kernel(const float* __restrict__ Yp,  // Np x cin
          const float* __restrict__ W,   // 8 x cin x cout2
          const int* __restrict__ par, const int* __restrict__ off,
          float* __restrict__ O,         // Nc x cout2
          int Nc, int cin, int cout2) {
  size_t i = (size_t)blockIdx.x * 256 + threadIdx.x;
  if (i >= (size_t)Nc * cout2) return;
  int r = (int)(i / cout2), co = (int)(i - (size_t)r * cout2);
  const float* w = W + ((size_t)off[r] * cin) * cout2 + co;
  const float* x = Yp + (size_t)par[r] * cin;
  float s = 0.f;
  for (int ci = 0; ci < cin; ++ci) s += x[ci] * w[(size_t)ci * cout2];
  O[i] = s;
}

// ---------------- concat [x, u] -> j (f32 + f16 mirror) ----------------
__global__ void __launch_bounds__(256)
concat_kernel(const float* __restrict__ A, const float* __restrict__ B,
              float* __restrict__ J, _Float16* __restrict__ Jh, int N, int p) {
  size_t i = (size_t)blockIdx.x * 256 + threadIdx.x;
  if (i >= (size_t)N * 2 * p) return;
  int r = (int)(i / (2 * p)), c = (int)(i - (size_t)r * 2 * p);
  float v = (c < p) ? A[(size_t)r * p + c] : B[(size_t)r * p + (c - p)];
  J[i] = v;
  Jh[i] = (_Float16)v;
}

// ---------------- x_new = j @ proj + h ----------------
__global__ void __launch_bounds__(256)
projadd_kernel(const float* __restrict__ J,  // N x 2p
               const float* __restrict__ P,  // 2p x p
               const float* __restrict__ H,  // N x p
               float* __restrict__ O, _Float16* __restrict__ Oh, int N, int p) {
  size_t i = (size_t)blockIdx.x * 256 + threadIdx.x;
  if (i >= (size_t)N * p) return;
  int r = (int)(i / p), c = (int)(i - (size_t)r * p);
  const float* j = J + (size_t)r * 2 * p;
  float s = H[i];
  for (int kk = 0; kk < 2 * p; ++kk) s += j[kk] * P[(size_t)kk * p + c];
  O[i] = s;
  Oh[i] = (_Float16)s;
}

// ---------------- final linear: x @ lin_W + lin_b ----------------
__global__ void __launch_bounds__(256)
linear_kernel(const float* __restrict__ X,  // N x 64
              const float* __restrict__ W,  // 64 x 32
              const float* __restrict__ b,
              float* __restrict__ out, int N) {
  size_t i = (size_t)blockIdx.x * 256 + threadIdx.x;
  if (i >= (size_t)N * 32) return;
  int r = (int)(i / 32), c = (int)(i - (size_t)r * 32);
  const float* x = X + (size_t)r * 64;
  float s = b[c];
  for (int kk = 0; kk < 64; ++kk) s += x[kk] * W[kk * 32 + c];
  out[i] = s;
}

// =====================================================================
extern "C" void kernel_launch(void* const* d_in, const int* in_sizes, int n_in,
                              void* d_out, int out_size, void* d_ws, size_t ws_size,
                              hipStream_t stream) {
  (void)n_in; (void)out_size; (void)ws_size;
  const int PL[4] = {64, 128, 256, 512};

  // ---- input index map (jax pytree: dicts flattened by sorted keys) ----
  // outer order: feats, params(leaves), nbr0..3, par0,off0,par1,off1,par2,off2
  // params leaves: levels[0..3], lin_W, lin_b, stem
  //   level l<3 (sorted): down{W,b,be,g} j1{...} j2{...} proj r1{...} r2{...} up{...} = 25
  //   level 3: r1{W,b,be,g} r2{W,b,be,g} = 8
  //   gc dict order: W(+0), b(+1), be(+2), g(+3)
  const int L3BASE = 76;                 // 1 + 3*25
  const int LINW = 84, LINB = 85;
  const int STEMW = 86, STEMBE = 88, STEMG = 89;
  const int NBR0 = 90, PAR0 = 94;

  int N[4];
  for (int l = 0; l < 4; ++l) N[l] = in_sizes[NBR0 + l] / 27;

  auto f32p = [&](int i) { return (const float*)d_in[i]; };
  auto i32p = [&](int i) { return (const int*)d_in[i]; };

  // ---- workspace bump allocator ----
  char* base = (char*)d_ws;
  size_t woff = 0;
  auto alloc = [&](size_t bytes) -> char* {
    size_t a = (woff + 255) & ~(size_t)255;
    woff = a + bytes;
    return base + a;
  };

  _Float16* Wt   = (_Float16*)alloc((size_t)27 * 512 * 1024 * sizeof(_Float16));
  float* scale   = (float*)alloc(1024 * sizeof(float));
  float* shift   = (float*)alloc(1024 * sizeof(float));
  _Float16* fH   = (_Float16*)alloc((size_t)N[0] * 32 * sizeof(_Float16));
  float* xF[4]; _Float16* xH[4];
  for (int l = 0; l < 4; ++l) {
    xF[l] = (float*)alloc((size_t)N[l] * PL[l] * sizeof(float));
    xH[l] = (_Float16*)alloc((size_t)N[l] * PL[l] * sizeof(_Float16));
  }
  float* jF[3]; _Float16* jH[3];
  for (int l = 0; l < 3; ++l) {
    jF[l] = (float*)alloc((size_t)N[l] * 2 * PL[l] * sizeof(float));
    jH[l] = (_Float16*)alloc((size_t)N[l] * 2 * PL[l] * sizeof(_Float16));
  }
  size_t T = 0;
  for (int l = 0; l < 4; ++l) { size_t t = (size_t)N[l] * 2 * PL[l]; if (t > T) T = t; }
  float* tA = (float*)alloc(T * sizeof(float));
  float* tB = (float*)alloc(T * sizeof(float));
  _Float16* tH = (_Float16*)alloc(T * sizeof(_Float16));

  auto grid1 = [](size_t n) { return dim3((unsigned)((n + 255) / 256)); };

  auto run_wtrans = [&](const float* W, int K, int cin, int cout2) {
    size_t n = (size_t)K * cin * cout2;
    wtrans_kernel<<<grid1(n), 256, 0, stream>>>(W, Wt, K, cin, cout2);
  };
  auto run_subm = [&](const _Float16* X, const int* nbr, float* Y, int n, int cin, int cout2) {
    switch (cout2 / 128) {
      case 1: {  // TPW=1, MT=4
        dim3 g((n + 63) / 64);
        size_t sh = (size_t)64 * cin * sizeof(_Float16) + 64 * 4;
        subm_wmma_kernel<1, 4><<<g, 256, sh, stream>>>(X, Wt, nbr, Y, n, cin, cout2);
      } break;
      case 2: {  // TPW=2, MT=2
        dim3 g((n + 31) / 32);
        size_t sh = (size_t)32 * cin * sizeof(_Float16) + 32 * 4;
        subm_wmma_kernel<2, 2><<<g, 256, sh, stream>>>(X, Wt, nbr, Y, n, cin, cout2);
      } break;
      case 4: {  // TPW=4, MT=2
        dim3 g((n + 31) / 32);
        size_t sh = (size_t)32 * cin * sizeof(_Float16) + 32 * 4;
        subm_wmma_kernel<4, 2><<<g, 256, sh, stream>>>(X, Wt, nbr, Y, n, cin, cout2);
      } break;
      default: {  // TPW=8, MT=1
        dim3 g((n + 15) / 16);
        size_t sh = (size_t)16 * cin * sizeof(_Float16) + 16 * 4;
        subm_wmma_kernel<8, 1><<<g, 256, sh, stream>>>(X, Wt, nbr, Y, n, cin, cout2);
      } break;
    }
  };
  auto run_stats = [&](const float* Y, int n, int C, const float* g, const float* be) {
    bn_stats_kernel<<<C / 8, 256, 0, stream>>>(Y, n, C, g, be, scale, shift);
  };
  auto run_gate = [&](const float* Y, const float* res, float* xf, _Float16* xh, int n, int p) {
    bn_gate_kernel<<<grid1((size_t)n * p), 256, 0, stream>>>(Y, scale, shift, res, xf, xh, n, p);
  };
  auto resblock = [&](int l, int r1, int r2) {
    int p = PL[l], c2 = 2 * p;
    run_wtrans(f32p(r1 + 0), 27, p, c2);
    run_subm(xH[l], i32p(NBR0 + l), tA, N[l], p, c2);
    run_stats(tA, N[l], c2, f32p(r1 + 3), f32p(r1 + 2));
    run_gate(tA, nullptr, tB, tH, N[l], p);
    run_wtrans(f32p(r2 + 0), 27, p, c2);
    run_subm(tH, i32p(NBR0 + l), tA, N[l], p, c2);
    run_stats(tA, N[l], c2, f32p(r2 + 3), f32p(r2 + 2));
    run_gate(tA, xF[l], xF[l], xH[l], N[l], p);  // residual add
  };

  // ---- stem ----
  tof16_kernel<<<grid1((size_t)N[0] * 32), 256, 0, stream>>>(f32p(0), fH, (size_t)N[0] * 32);
  run_wtrans(f32p(STEMW), 27, 32, 128);
  run_subm(fH, i32p(NBR0), tA, N[0], 32, 128);
  run_stats(tA, N[0], 128, f32p(STEMG), f32p(STEMBE));
  run_gate(tA, nullptr, xF[0], xH[0], N[0], 64);

  // ---- encoder ----
  for (int l = 0; l < 3; ++l) {
    int b = 1 + 25 * l;
    resblock(l, b + 13, b + 17);
    int pn = PL[l + 1], c2n = 2 * pn;
    zero_kernel<<<grid1((size_t)N[l + 1] * c2n), 256, 0, stream>>>(tA, (size_t)N[l + 1] * c2n);
    down_kernel<<<grid1((size_t)N[l] * c2n), 256, 0, stream>>>(
        xF[l], f32p(b + 0), i32p(PAR0 + 2 * l), i32p(PAR0 + 2 * l + 1), tA, N[l], PL[l], c2n);
    run_stats(tA, N[l + 1], c2n, f32p(b + 3), f32p(b + 2));
    run_gate(tA, nullptr, xF[l + 1], xH[l + 1], N[l + 1], pn);
  }

  // ---- bottom ----
  resblock(3, L3BASE, L3BASE + 4);

  // ---- decoder ----
  for (int l = 2; l >= 0; --l) {
    int b = 1 + 25 * l;
    int p = PL[l], c2 = 2 * p, pn = PL[l + 1];
    up_kernel<<<grid1((size_t)N[l] * c2), 256, 0, stream>>>(
        xF[l + 1], f32p(b + 21), i32p(PAR0 + 2 * l), i32p(PAR0 + 2 * l + 1), tA, N[l], pn, c2);
    run_stats(tA, N[l], c2, f32p(b + 24), f32p(b + 23));
    run_gate(tA, nullptr, tB, tH, N[l], p);  // u in tB
    concat_kernel<<<grid1((size_t)N[l] * c2), 256, 0, stream>>>(xF[l], tB, jF[l], jH[l], N[l], p);
    run_wtrans(f32p(b + 4), 27, c2, c2);                    // j1: cin=2p, cout2=2p
    run_subm(jH[l], i32p(NBR0 + l), tA, N[l], c2, c2);
    run_stats(tA, N[l], c2, f32p(b + 7), f32p(b + 6));
    run_gate(tA, nullptr, tB, tH, N[l], p);
    run_wtrans(f32p(b + 8), 27, p, c2);                     // j2: cin=p, cout2=2p
    run_subm(tH, i32p(NBR0 + l), tA, N[l], p, c2);
    run_stats(tA, N[l], c2, f32p(b + 11), f32p(b + 10));
    run_gate(tA, nullptr, tB, tH, N[l], p);                 // h in tB
    projadd_kernel<<<grid1((size_t)N[l] * p), 256, 0, stream>>>(
        jF[l], f32p(b + 12), tB, xF[l], xH[l], N[l], p);
  }

  // ---- head ----
  linear_kernel<<<grid1((size_t)N[0] * 32), 256, 0, stream>>>(
      xF[0], f32p(LINW), f32p(LINB), (float*)d_out, N[0]);
}
